// GCN_78219944394959
// MI455X (gfx1250) — matrix operand order repaired and checked by reading, not compiled, not run on hardware
//
#include <hip/hip_runtime.h>
#include <hip/hip_bf16.h>
#include <math.h>

#define N_NODES 50000
#define N_EDGES 800000
#define IN_FEAT 512
#define HID     256
#define OUT_F   64

typedef __attribute__((ext_vector_type(16))) __bf16 v16bf;
typedef __attribute__((ext_vector_type(8)))  __bf16 v8bf;
typedef __attribute__((ext_vector_type(8)))  float  v8f;

__device__ __forceinline__ unsigned short f2bfu(float f) {
    union { float f; unsigned u; } a; a.f = f;
    unsigned u = a.u;
    return (unsigned short)((u + 0x7FFFu + ((u >> 16) & 1u)) >> 16); // RNE
}

// ---------------------------------------------------------------------------
// Kernel 0: CSR row pointers from sorted COO rows (lower_bound binary search)
// ---------------------------------------------------------------------------
__global__ void rowptr_kernel(const int* __restrict__ a_row, int* __restrict__ rp) {
    int r = blockIdx.x * blockDim.x + threadIdx.x;
    if (r > N_NODES) return;
    int lo = 0, hi = N_EDGES;
    while (lo < hi) {
        int mid = (lo + hi) >> 1;
        if (a_row[mid] < r) lo = mid + 1; else hi = mid;
    }
    rp[r] = lo;
}

// ---------------------------------------------------------------------------
// Kernel 1: X (f32) -> Xbf (bf16 row-major). 4 elements/thread, 8B stores.
// ---------------------------------------------------------------------------
__global__ __launch_bounds__(256) void cvt_x_kernel(const float* __restrict__ X,
                                                    unsigned short* __restrict__ Xbf) {
    const size_t base = ((size_t)blockIdx.x * 256 + threadIdx.x) * 4;
    if (base >= (size_t)N_NODES * IN_FEAT) return;
    const float4 v = *(const float4*)(X + base);
    ushort4 o;
    o.x = f2bfu(v.x); o.y = f2bfu(v.y); o.z = f2bfu(v.z); o.w = f2bfu(v.w);
    *(ushort4*)(Xbf + base) = o;
}

// ---------------------------------------------------------------------------
// Kernel 2: pack W (K x Ncols, f32 row-major) into B-fragment-major bf16:
//   Wp[((kt*2+half)*Ncols + n)*16 + j] = bf16(W[(kt*32 + half*16 + j)*Ncols + n])
// One lane's 16 K-values for a WMMA B fragment are 32 contiguous bytes, and
// adjacent 16-col tiles sit +512B apart (immediate-offset loadable).
// ---------------------------------------------------------------------------
__global__ __launch_bounds__(256) void pack_w_kernel(const float* __restrict__ W,
                                                     unsigned short* __restrict__ Wp,
                                                     int K, int Ncols) {
    const int idx = blockIdx.x * 256 + threadIdx.x;
    if (idx >= K * Ncols) return;
    const int j = idx & 15;
    const int n = (idx >> 4) % Ncols;
    const int g = (idx >> 4) / Ncols;               // g = kt*2 + half
    const int k = (g >> 1) * 32 + (g & 1) * 16 + j;
    Wp[idx] = f2bfu(W[(size_t)k * Ncols + n]);
}

// ---------------------------------------------------------------------------
// GEMM core, dual M-tile, pointer-walking (K/Ncols compile-time so all strides
// are literals): per K-step = 4 A-loads + 8 B-loads (immediate offsets off one
// base) + 8 WMMA. Each B fragment feeds two WMMAs.
// ---------------------------------------------------------------------------
__device__ __forceinline__ v16bf cat8(v8bf lo, v8bf hi) {
    return __builtin_shufflevector(lo, hi,
        0, 1, 2, 3, 4, 5, 6, 7, 8, 9, 10, 11, 12, 13, 14, 15);
}

template <int K, int NCOLS>
__device__ __forceinline__ void wmma_dual_slab(const unsigned short* __restrict__ arow0,
                                               const unsigned short* __restrict__ arow1,
                                               const unsigned short* __restrict__ Wp,
                                               int c0, int half, int l16,
                                               v8f acc0[4], v8f acc1[4]) {
    const __bf16* ap0 = (const __bf16*)arow0 + half * 16;
    const __bf16* ap1 = (const __bf16*)arow1 + half * 16;
    const __bf16* bp  = (const __bf16*)Wp + ((size_t)half * NCOLS + c0 + l16) * 16;
    for (int k0 = 0; k0 < K; k0 += 32) {
        const v16bf a0 = cat8(*(const v8bf*)ap0, *(const v8bf*)(ap0 + 8));
        const v16bf a1 = cat8(*(const v8bf*)ap1, *(const v8bf*)(ap1 + 8));
        __builtin_prefetch(ap0 + 32, 0, 3);            // global_prefetch_b8
        __builtin_prefetch(ap1 + 32, 0, 3);
#pragma unroll
        for (int ct = 0; ct < 4; ++ct) {
            const v16bf b = *(const v16bf*)(bp + ct * 256);  // +512B per tile
            acc0[ct] = __builtin_amdgcn_wmma_f32_16x16x32_bf16(
                false, a0, false, b, (short)0, acc0[ct], false, false);
            acc1[ct] = __builtin_amdgcn_wmma_f32_16x16x32_bf16(
                false, a1, false, b, (short)0, acc1[ct], false, false);
        }
        ap0 += 32;
        ap1 += 32;
        bp  += (size_t)2 * NCOLS * 16;                 // literal stride
    }
}

__device__ __forceinline__ void store_ctile(float* __restrict__ C, int ldc, int r0,
                                            int c0, int half, int l16, const v8f acc[4]) {
    // C/D layout: VGPR v -> M=v (lanes 0-15) / M=8+v (lanes 16-31), N=l16.
#pragma unroll
    for (int ct = 0; ct < 4; ++ct) {
        const int n = c0 + ct * 16 + l16;
#pragma unroll
        for (int v = 0; v < 8; ++v) {
            const int row = r0 + v + 8 * half;
            if (row < N_NODES) C[(size_t)row * ldc + n] = acc[ct][v];
        }
    }
}

// ---------------------------------------------------------------------------
// Kernel 3: H0 = Xbf @ W1p. Block = 256 threads (8 waves): wave -> 32-row
// pair (w>>2), 64-col slab (w&3). Block covers 64 rows x 256 cols.
// ---------------------------------------------------------------------------
__global__ __launch_bounds__(256) void gemm1_wmma(const unsigned short* __restrict__ Xbf,
                                                  const unsigned short* __restrict__ W1p,
                                                  float* __restrict__ H0) {
    const int wave = threadIdx.x >> 5;
    const int lane = threadIdx.x & 31;
    const int half = lane >> 4;
    const int l16  = lane & 15;

    const int r0 = (blockIdx.x * 4 + (wave >> 2) * 2) * 16;   // first 16-row tile
    const int c0 = (wave & 3) * 64;
    if (r0 >= N_NODES) return;

    int arow0 = r0 + l16;      if (arow0 >= N_NODES) arow0 = N_NODES - 1;
    int arow1 = r0 + 16 + l16; if (arow1 >= N_NODES) arow1 = N_NODES - 1;

    v8f acc0[4] = {}, acc1[4] = {};
    wmma_dual_slab<IN_FEAT, HID>(Xbf + (size_t)arow0 * IN_FEAT,
                                 Xbf + (size_t)arow1 * IN_FEAT,
                                 W1p, c0, half, l16, acc0, acc1);

    store_ctile(H0, HID, r0,      c0, half, l16, acc0);
    store_ctile(H0, HID, r0 + 16, c0, half, l16, acc1);
}

// ---------------------------------------------------------------------------
// Kernel 4: h = relu(A @ H0) * drop_mask * 2, bf16 out. One 64-thread block
// per row; each lane owns 4 contiguous features (float4 gathers, L2-resident).
// ---------------------------------------------------------------------------
__global__ __launch_bounds__(64) void spmm1_kernel(const int* __restrict__ rp,
                                                   const int* __restrict__ a_col,
                                                   const float* __restrict__ a_val,
                                                   const float* __restrict__ H0,
                                                   const int* __restrict__ drop,
                                                   unsigned short* __restrict__ hbf) {
    const int r  = blockIdx.x;
    const int f4 = threadIdx.x * 4;
    const int e0 = rp[r], e1 = rp[r + 1];
    float ax = 0.f, ay = 0.f, az = 0.f, aw = 0.f;
    for (int e = e0; e < e1; ++e) {
        const int c = a_col[e];
        const float w = a_val[e];
        const float4 h4 = *(const float4*)(H0 + (size_t)c * HID + f4);
        ax = fmaf(w, h4.x, ax); ay = fmaf(w, h4.y, ay);
        az = fmaf(w, h4.z, az); aw = fmaf(w, h4.w, aw);
    }
    const int4 m = *(const int4*)(drop + (size_t)r * HID + f4);
    ushort4 o;
    o.x = f2bfu(fmaxf(ax, 0.f) * (float)m.x * 2.0f);
    o.y = f2bfu(fmaxf(ay, 0.f) * (float)m.y * 2.0f);
    o.z = f2bfu(fmaxf(az, 0.f) * (float)m.z * 2.0f);
    o.w = f2bfu(fmaxf(aw, 0.f) * (float)m.w * 2.0f);
    *(ushort4*)(hbf + (size_t)r * HID + f4) = o;
}

// ---------------------------------------------------------------------------
// Kernel 5: H2 = hbf @ W2p. Block = 256 threads (8 waves); each wave owns
// TWO 16-row tiles x all 64 output cols. Block covers 256 rows. K = 256.
// ---------------------------------------------------------------------------
__global__ __launch_bounds__(256) void gemm2_wmma(const unsigned short* __restrict__ Hbf,
                                                  const unsigned short* __restrict__ W2p,
                                                  float* __restrict__ H2) {
    const int wave = threadIdx.x >> 5;
    const int lane = threadIdx.x & 31;
    const int half = lane >> 4;
    const int l16  = lane & 15;

    const int r0 = (blockIdx.x * 16 + wave * 2) * 16;
    if (r0 >= N_NODES) return;

    int arow0 = r0 + l16;      if (arow0 >= N_NODES) arow0 = N_NODES - 1;
    int arow1 = r0 + 16 + l16; if (arow1 >= N_NODES) arow1 = N_NODES - 1;

    v8f acc0[4] = {}, acc1[4] = {};
    wmma_dual_slab<HID, OUT_F>(Hbf + (size_t)arow0 * HID,
                               Hbf + (size_t)arow1 * HID,
                               W2p, 0, half, l16, acc0, acc1);

    store_ctile(H2, OUT_F, r0,      0, half, l16, acc0);
    store_ctile(H2, OUT_F, r0 + 16, 0, half, l16, acc1);
}

// ---------------------------------------------------------------------------
// Kernel 6: out = log_softmax(A @ H2, axis=1). One 64-thread block per row.
// ---------------------------------------------------------------------------
__global__ __launch_bounds__(64) void spmm2_lsm_kernel(const int* __restrict__ rp,
                                                       const int* __restrict__ a_col,
                                                       const float* __restrict__ a_val,
                                                       const float* __restrict__ H2,
                                                       float* __restrict__ out) {
    __shared__ float red[64];
    const int r = blockIdx.x;
    const int t = threadIdx.x;
    const int e0 = rp[r], e1 = rp[r + 1];
    float acc = 0.0f;
    for (int e = e0; e < e1; ++e)
        acc = fmaf(a_val[e], H2[(size_t)a_col[e] * OUT_F + t], acc);

    red[t] = acc; __syncthreads();
#pragma unroll
    for (int s = 32; s > 0; s >>= 1) {
        if (t < s) red[t] = fmaxf(red[t], red[t + s]);
        __syncthreads();
    }
    const float mx = red[0];
    __syncthreads();

    red[t] = __expf(acc - mx); __syncthreads();
#pragma unroll
    for (int s = 32; s > 0; s >>= 1) {
        if (t < s) red[t] += red[t + s];
        __syncthreads();
    }
    const float lse = __logf(red[0]);
    out[(size_t)r * OUT_F + t] = (acc - mx) - lse;
}

// ---------------------------------------------------------------------------
// Launcher. Workspace layout (bytes, 256-aligned):
//   Xbf bf16 [50000,512] @ 0            (51,200,000)
//   H0  f32  [50000,256] @  51,200,000  (51,200,000)
//   hbf bf16 [50000,256] @ 102,400,000  (25,600,000)
//   H2  f32  [50000, 64] @ 128,000,000  (12,800,000)
//   W1p bf16 [512,256]   @ 140,800,000  (   262,144)
//   W2p bf16 [256, 64]   @ 141,062,400  (    32,768)
//   rp  int  [50001]     @ 141,095,168  (   200,004)
// ---------------------------------------------------------------------------
extern "C" void kernel_launch(void* const* d_in, const int* in_sizes, int n_in,
                              void* d_out, int out_size, void* d_ws, size_t ws_size,
                              hipStream_t stream) {
    const float* X     = (const float*)d_in[0];
    const float* W1    = (const float*)d_in[1];
    const float* W2    = (const float*)d_in[2];
    const int*   a_row = (const int*)d_in[3];
    const int*   a_col = (const int*)d_in[4];
    const float* a_val = (const float*)d_in[5];
    const int*   drop  = (const int*)d_in[6];
    float*       out   = (float*)d_out;

    char* ws = (char*)d_ws;
    unsigned short* Xbf = (unsigned short*)(ws + 0);
    float*          H0  = (float*)(ws + 51200000);
    unsigned short* hbf = (unsigned short*)(ws + 102400000);
    float*          H2  = (float*)(ws + 128000000);
    unsigned short* W1p = (unsigned short*)(ws + 140800000);
    unsigned short* W2p = (unsigned short*)(ws + 141062400);
    int*            rp  = (int*)(ws + 141095168);

    rowptr_kernel<<<(N_NODES + 1 + 255) / 256, 256, 0, stream>>>(a_row, rp);
    cvt_x_kernel<<<(N_NODES * IN_FEAT / 4 + 255) / 256, 256, 0, stream>>>(X, Xbf);
    pack_w_kernel<<<(IN_FEAT * HID + 255) / 256, 256, 0, stream>>>(W1, W1p, IN_FEAT, HID);
    pack_w_kernel<<<(HID * OUT_F + 255) / 256, 256, 0, stream>>>(W2, W2p, HID, OUT_F);

    gemm1_wmma<<<(N_NODES + 63) / 64, 256, 0, stream>>>(Xbf, W1p, H0);
    spmm1_kernel<<<N_NODES, 64, 0, stream>>>(rp, a_col, a_val, H0, drop, hbf);
    gemm2_wmma<<<(N_NODES + 255) / 256, 256, 0, stream>>>(hbf, W2p, H2);
    spmm2_lsm_kernel<<<N_NODES, 64, 0, stream>>>(rp, a_col, a_val, H2, out);
}
